// GRU_51908974739555
// MI455X (gfx1250) — matrix operand order repaired and checked by reading, not compiled, not run on hardware
//
#include <hip/hip_runtime.h>
#include <hip/hip_bf16.h>

// ---------------------------------------------------------------------------
// Types for CDNA5 WMMA (wave32): v_wmma_f32_16x16x32_bf16
// ---------------------------------------------------------------------------
typedef __attribute__((ext_vector_type(16))) __bf16          v16bf;
typedef __attribute__((ext_vector_type(8)))  float           v8f;
typedef __attribute__((ext_vector_type(16))) unsigned short  v16us;
typedef __attribute__((ext_vector_type(8)))  unsigned short  v8us;

union Frag16 { v16us us; v16bf bf; };

// f32 -> bf16 round-to-nearest-even (bit pattern)
__device__ __forceinline__ unsigned short f2bf(float f) {
    unsigned int u = __float_as_uint(f);
    u += 0x7FFFu + ((u >> 16) & 1u);
    return (unsigned short)(u >> 16);
}

// Build a 16-element bf16 fragment from two 16-byte LDS segments.
__device__ __forceinline__ v16bf ld_frag(const unsigned short* p0,
                                         const unsigned short* p1) {
    Frag16 f;
    v8us a = *(const v8us*)p0;
    v8us b = *(const v8us*)p1;
#pragma unroll
    for (int i = 0; i < 8; ++i) { f.us[i] = a[i]; f.us[i + 8] = b[i]; }
    return f.bf;
}

// Load a WMMA B-fragment of W[192][64] (row-major f32) for column tile row
// `n`, contiguous K range starting at kbase (16 values), converted to bf16.
__device__ __forceinline__ v16bf load_wfrag(const float* __restrict__ W,
                                            int n, int kbase) {
    Frag16 f;
    const float* p = W + n * 64 + kbase;
#pragma unroll
    for (int i = 0; i < 16; ++i) f.us[i] = f2bf(p[i]);
    return f.bf;
}

// Fast sigmoid: one v_exp + one v_rcp (no IEEE divide sequence).
__device__ __forceinline__ float sigmoid_fast(float x) {
    return __builtin_amdgcn_rcpf(1.0f + __expf(-x));
}

// Fast tanh: tanh(x) = sign(x) * (1 - e) / (1 + e), e = exp(-2|x|).
__device__ __forceinline__ float tanh_fast(float x) {
    const float e = __expf(-2.0f * __builtin_fabsf(x));
    const float y = (1.0f - e) * __builtin_amdgcn_rcpf(1.0f + e);
    return __builtin_copysignf(y, x);
}

// ---------------------------------------------------------------------------
// Generic GEMM: C[M,N] = act(A[M,K] * W[N,K]^T + bias), bf16 WMMA, f32 accum.
// Block: 256 threads (8 waves). Each wave owns 16 rows of M; block M-tile 128.
// NTILES = ceil(N/16) (templated so accumulators stay in registers).
// ---------------------------------------------------------------------------
template <int NTILES>
__global__ __launch_bounds__(256)
void gemm_bf16(const float* __restrict__ A, const float* __restrict__ W,
               const float* __restrict__ bias, float* __restrict__ C,
               int N, int K, int relu) {
    __shared__ unsigned short sA[128 * 32];          // [128][32] bf16
    __shared__ unsigned short sB[NTILES * 16 * 32];  // [N_pad][32] bf16

    const int tid  = threadIdx.x;
    const int wave = tid >> 5;
    const int lane = tid & 31;
    const int l16  = lane & 15;
    const int hi   = lane >> 4;          // 0 or 1 (lane half)
    const long row0 = (long)blockIdx.x * 128;

    v8f acc[NTILES];
#pragma unroll
    for (int t = 0; t < NTILES; ++t)
#pragma unroll
        for (int j = 0; j < 8; ++j) acc[t][j] = 0.0f;

    const int ksteps = (K + 31) >> 5;
    for (int ks = 0; ks < ksteps; ++ks) {
        const int k0 = ks << 5;
        __syncthreads();
        // Stage A tile 128x32 (f32 -> bf16), coalesced over lanes.
#pragma unroll
        for (int i = 0; i < 16; ++i) {
            int e = tid + i * 256;               // 0..4095
            int r = e >> 5, c = e & 31;
            float v = (k0 + c < K) ? A[(row0 + r) * (long)K + k0 + c] : 0.0f;
            sA[e] = f2bf(v);
        }
        // Stage B tile (N_pad x 32).
        for (int e = tid; e < NTILES * 16 * 32; e += 256) {
            int r = e >> 5, c = e & 31;
            float v = (r < N && k0 + c < K) ? W[(long)r * K + k0 + c] : 0.0f;
            sB[e] = f2bf(v);
        }
        __syncthreads();

        // A fragment for this wave's 16 rows.
        const int am = wave * 16 + l16;
        v16bf afrag = ld_frag(&sA[am * 32 + (hi ? 16 : 0)],
                              &sA[am * 32 + (hi ? 24 : 8)]);
#pragma unroll
        for (int nt = 0; nt < NTILES; ++nt) {
            const int bn = nt * 16 + l16;
            v16bf bfrag = ld_frag(&sB[bn * 32 + (hi ? 16 : 0)],
                                  &sB[bn * 32 + (hi ? 16 : 0) + 8]);
            acc[nt] = __builtin_amdgcn_wmma_f32_16x16x32_bf16(
                false, afrag, false, bfrag, (short)0, acc[nt], false, false);
        }
    }

    // C layout: vgpr i, lane L -> row = i + 8*(L>=16), col = L%16.
    const int mrow = wave * 16 + hi * 8;
    const int col  = l16;
#pragma unroll
    for (int nt = 0; nt < NTILES; ++nt) {
        const int n = nt * 16 + col;
        if (n < N) {
            const float bv = bias[n];
#pragma unroll
            for (int i = 0; i < 8; ++i) {
                float v = acc[nt][i] + bv;
                if (relu) v = fmaxf(v, 0.0f);
                C[(row0 + mrow + i) * (long)N + n] = v;
            }
        }
    }
}

// ---------------------------------------------------------------------------
// Fused 3-layer GRU recurrence. Grid = 16 blocks (batch slices of 16 rows),
// block = 384 threads (12 waves, one per 16-wide N tile of the 192 gates).
// Hidden states live in LDS (f32 + bf16 copies); the five recurrent weight
// matrices are held in registers as pre-converted WMMA B-fragments.
// xg0 = x*w_ih0^T + b_ih0 is precomputed by gemm_bf16<12>.
// Gate order r,z,n (PyTorch): n = tanh(gi_n + r*(hg_n + b_hh_n)).
// ---------------------------------------------------------------------------
__global__ __launch_bounds__(384)
void gru3_recurrent(const float* __restrict__ xg0,   // [B,T,192] (bias included)
                    const float* __restrict__ whh0, const float* __restrict__ bhh0,
                    const float* __restrict__ wih1, const float* __restrict__ whh1,
                    const float* __restrict__ bih1, const float* __restrict__ bhh1,
                    const float* __restrict__ wih2, const float* __restrict__ whh2,
                    const float* __restrict__ bih2, const float* __restrict__ bhh2,
                    float* __restrict__ out,          // [B,T,64] = relu(h2)
                    int T) {
    __shared__ unsigned short h0b[16 * 64], h1b[16 * 64], h2b[16 * 64];
    __shared__ float h0f[16 * 64], h1f[16 * 64], h2f[16 * 64];
    __shared__ float xgb[16 * 192], hgb[16 * 192];

    const int tid  = threadIdx.x;
    const int wave = tid >> 5;           // 0..11 == gate-column tile
    const int lane = tid & 31;
    const int l16  = lane & 15;
    const int hi   = lane >> 4;
    const int b0   = blockIdx.x * 16;

    // Resident weight fragments (B side of h[16,64] @ W[192,64]^T).
    v16bf Whh0[2], Wih1[2], Whh1[2], Wih2[2], Whh2[2];
#pragma unroll
    for (int ks = 0; ks < 2; ++ks) {
        const int n = wave * 16 + l16;
        const int kb = ks * 32 + hi * 16;
        Whh0[ks] = load_wfrag(whh0, n, kb);
        Wih1[ks] = load_wfrag(wih1, n, kb);
        Whh1[ks] = load_wfrag(whh1, n, kb);
        Wih2[ks] = load_wfrag(wih2, n, kb);
        Whh2[ks] = load_wfrag(whh2, n, kb);
    }

    // Zero-init hidden states.
    for (int e = tid; e < 16 * 64; e += 384) {
        h0f[e] = 0.0f; h1f[e] = 0.0f; h2f[e] = 0.0f;
        h0b[e] = 0;    h1b[e] = 0;    h2b[e] = 0;
    }
    __syncthreads();

    // One M16 x N192 x K64 WMMA GEMM (this wave's N-tile) from bf16 LDS state.
    auto gemm_tile = [&](const unsigned short* hb, const v16bf* wf,
                         float* dst) {
        v8f g;
#pragma unroll
        for (int j = 0; j < 8; ++j) g[j] = 0.0f;
#pragma unroll
        for (int ks = 0; ks < 2; ++ks) {
            v16bf a = ld_frag(&hb[l16 * 64 + ks * 32 + (hi ? 16 : 0)],
                              &hb[l16 * 64 + ks * 32 + (hi ? 24 : 8)]);
            g = __builtin_amdgcn_wmma_f32_16x16x32_bf16(
                false, a, false, wf[ks], (short)0, g, false, false);
        }
#pragma unroll
        for (int i = 0; i < 8; ++i)
            dst[(i + hi * 8) * 192 + wave * 16 + l16] = g[i];
    };

    for (int t = 0; t < T; ++t) {
        // ----- layer 0: hg0 = h0 @ whh0^T -----
        gemm_tile(h0b, Whh0, hgb);
        __syncthreads();
#pragma unroll
        for (int it = 0; it < 3; ++it) {
            const int e = tid + it * 384;
            if (e < 16 * 64) {
                const int b = e >> 6, j = e & 63;
                const long base = ((long)(b0 + b) * T + t) * 192;
                const float gr = xg0[base + j];
                const float gz = xg0[base + 64 + j];
                const float gn = xg0[base + 128 + j];
                const float hr = hgb[b * 192 + j]       + bhh0[j];
                const float hz = hgb[b * 192 + 64 + j]  + bhh0[64 + j];
                const float hn = hgb[b * 192 + 128 + j] + bhh0[128 + j];
                const float r = sigmoid_fast(gr + hr);
                const float z = sigmoid_fast(gz + hz);
                const float n = tanh_fast(gn + r * hn);
                const float h = (1.0f - z) * n + z * h0f[e];
                h0f[e] = h; h0b[e] = f2bf(h);
            }
        }
        __syncthreads();

        // ----- layer 1: xg1 = h0new @ wih1^T ; hg1 = h1 @ whh1^T -----
        gemm_tile(h0b, Wih1, xgb);
        gemm_tile(h1b, Whh1, hgb);
        __syncthreads();
#pragma unroll
        for (int it = 0; it < 3; ++it) {
            const int e = tid + it * 384;
            if (e < 16 * 64) {
                const int b = e >> 6, j = e & 63;
                const float gr = xgb[b * 192 + j]       + bih1[j];
                const float gz = xgb[b * 192 + 64 + j]  + bih1[64 + j];
                const float gn = xgb[b * 192 + 128 + j] + bih1[128 + j];
                const float hr = hgb[b * 192 + j]       + bhh1[j];
                const float hz = hgb[b * 192 + 64 + j]  + bhh1[64 + j];
                const float hn = hgb[b * 192 + 128 + j] + bhh1[128 + j];
                const float r = sigmoid_fast(gr + hr);
                const float z = sigmoid_fast(gz + hz);
                const float n = tanh_fast(gn + r * hn);
                const float h = (1.0f - z) * n + z * h1f[e];
                h1f[e] = h; h1b[e] = f2bf(h);
            }
        }
        __syncthreads();

        // ----- layer 2: xg2 = h1new @ wih2^T ; hg2 = h2 @ whh2^T -----
        gemm_tile(h1b, Wih2, xgb);
        gemm_tile(h2b, Whh2, hgb);
        __syncthreads();
#pragma unroll
        for (int it = 0; it < 3; ++it) {
            const int e = tid + it * 384;
            if (e < 16 * 64) {
                const int b = e >> 6, j = e & 63;
                const float gr = xgb[b * 192 + j]       + bih2[j];
                const float gz = xgb[b * 192 + 64 + j]  + bih2[64 + j];
                const float gn = xgb[b * 192 + 128 + j] + bih2[128 + j];
                const float hr = hgb[b * 192 + j]       + bhh2[j];
                const float hz = hgb[b * 192 + 64 + j]  + bhh2[64 + j];
                const float hn = hgb[b * 192 + 128 + j] + bhh2[128 + j];
                const float r = sigmoid_fast(gr + hr);
                const float z = sigmoid_fast(gz + hz);
                const float n = tanh_fast(gn + r * hn);
                const float h = (1.0f - z) * n + z * h2f[e];
                h2f[e] = h; h2b[e] = f2bf(h);
                out[((long)(b0 + b) * T + t) * 64 + j] = fmaxf(h, 0.0f);
            }
        }
        __syncthreads();
    }
}

// ---------------------------------------------------------------------------
// Host-side orchestration.
// ---------------------------------------------------------------------------
extern "C" void kernel_launch(void* const* d_in, const int* in_sizes, int n_in,
                              void* d_out, int out_size, void* d_ws, size_t ws_size,
                              hipStream_t stream) {
    (void)in_sizes; (void)n_in; (void)out_size; (void)ws_size;
    constexpr int B = 256, T = 384, D = 1629, H = 64, NCLS = 250;
    constexpr long M = (long)B * T;                 // 98304 rows

    const float* x     = (const float*)d_in[0];
    const float* wih0  = (const float*)d_in[1];
    const float* whh0  = (const float*)d_in[2];
    const float* bih0  = (const float*)d_in[3];
    const float* bhh0  = (const float*)d_in[4];
    const float* wih1  = (const float*)d_in[5];
    const float* whh1  = (const float*)d_in[6];
    const float* bih1  = (const float*)d_in[7];
    const float* bhh1  = (const float*)d_in[8];
    const float* wih2  = (const float*)d_in[9];
    const float* whh2  = (const float*)d_in[10];
    const float* bih2  = (const float*)d_in[11];
    const float* bhh2  = (const float*)d_in[12];
    const float* w1    = (const float*)d_in[13];
    const float* b1    = (const float*)d_in[14];
    const float* w2    = (const float*)d_in[15];
    const float* b2    = (const float*)d_in[16];
    const float* w3    = (const float*)d_in[17];
    const float* b3    = (const float*)d_in[18];

    float* ws  = (float*)d_ws;
    float* xg0 = ws;                       // M * 192
    float* h2  = xg0 + M * 192;            // M * 64 (relu'd GRU output)
    float* a1  = h2  + M * 64;             // M * 64
    float* a2  = a1  + M * 64;             // M * 32
    float* outp = (float*)d_out;           // M * 250

    const int grid = (int)(M / 128);       // 768

    // 1) big input projection: xg0 = x @ wih0^T + bih0
    gemm_bf16<12><<<grid, 256, 0, stream>>>(x, wih0, bih0, xg0, 3 * H, D, 0);

    // 2) fused 3-layer GRU recurrence -> relu(h2) sequence
    gru3_recurrent<<<B / 16, 384, 0, stream>>>(xg0, whh0, bhh0,
                                               wih1, whh1, bih1, bhh1,
                                               wih2, whh2, bih2, bhh2,
                                               h2, T);

    // 3) MLP head (all relu'd)
    gemm_bf16<4><<<grid, 256, 0, stream>>>(h2, w1, b1, a1, H, H, 1);
    gemm_bf16<2><<<grid, 256, 0, stream>>>(a1, w2, b2, a2, 32, H, 1);
    gemm_bf16<16><<<grid, 256, 0, stream>>>(a2, w3, b3, outp, NCLS, 32, 1);
}
